// RNNEncoder_30554397344018
// MI455X (gfx1250) — matrix operand order repaired and checked by reading, not compiled, not run on hardware
//
#include <hip/hip_runtime.h>

typedef __bf16 bf16;
typedef __attribute__((ext_vector_type(16))) __bf16 v16bf;
typedef __attribute__((ext_vector_type(8)))  __bf16 v8bf;
typedef __attribute__((ext_vector_type(8)))  float  v8f;

#define EPSV 1e-5f

__device__ __forceinline__ float sigf(float x) { return 1.0f / (1.0f + __expf(-x)); }

// ---------------------------------------------------------------------------
// fp32 -> bf16 conversion (prep)
// ---------------------------------------------------------------------------
__global__ void __launch_bounds__(256)
k_cvt(const float* __restrict__ s, bf16* __restrict__ d, int n) {
    for (int i = blockIdx.x * 256 + threadIdx.x; i < n; i += gridDim.x * 256)
        d[i] = (bf16)s[i];
}

// ---------------------------------------------------------------------------
// Phase 1: Gx = LN(x @ Wih^T + bih) for both directions.
// grid (2048, 2): blockIdx.x = M-tile (16 rows of B*T=32768), blockIdx.y = dir
// block 256 = 8 waves; wave w owns N-tiles [12w, 12w+12) of 96 (N = 1536)
// ---------------------------------------------------------------------------
__global__ void __launch_bounds__(256)
k_inproj(const bf16* __restrict__ xb,
         const bf16* __restrict__ WihF, const bf16* __restrict__ WihB,
         const float* __restrict__ bihF, const float* __restrict__ bihB,
         const float* __restrict__ gF,  const float* __restrict__ beF,
         const float* __restrict__ gB,  const float* __restrict__ beB,
         bf16* __restrict__ GxF, bf16* __restrict__ GxB)
{
    const int dir = blockIdx.y;
    const bf16*  W   = dir ? WihB : WihF;
    const float* bih = dir ? bihB : bihF;
    const float* gg  = dir ? gB  : gF;
    const float* be  = dir ? beB : beF;
    bf16* Gx         = dir ? GxB : GxF;

    const int m0   = blockIdx.x * 16;
    const int tid  = threadIdx.x;
    const int lane = tid & 31;
    const int wv   = tid >> 5;
    const int hs   = lane >> 4;     // half-select (0: lanes 0-15, 1: lanes 16-31)
    const int l15  = lane & 15;

    __shared__ bf16  a_s[16 * 784];       // A tile, padded stride
    __shared__ bf16  o_s[16 * 1544];      // output staging, padded stride
    __shared__ float rp[8][16], rp2[8][16];

    // stage A tile: 16 rows x 768 bf16, 16B chunks
    for (int idx = tid; idx < 16 * 96; idx += 256) {
        int r = idx / 96, c = idx % 96;
        *(uint4*)&a_s[r * 784 + c * 8] =
            *(const uint4*)&xb[(size_t)(m0 + r) * 768 + c * 8];
    }
    __syncthreads();

    // per-lane base pointer into W for this wave's 12 N-tiles
    const bf16* Wbase = W + (size_t)((wv * 12) * 16 + l15) * 768 + (hs ? 16 : 0);

    v8f acc[12];
#pragma unroll
    for (int nt = 0; nt < 12; ++nt)
#pragma unroll
        for (int i = 0; i < 8; ++i) acc[nt][i] = 0.0f;

    for (int kt = 0; kt < 24; ++kt) {
        const int kb = kt * 32;
        v8bf a0 = *(const v8bf*)&a_s[l15 * 784 + kb + (hs ? 16 : 0)];
        v8bf a1 = *(const v8bf*)&a_s[l15 * 784 + kb + (hs ? 16 : 0) + 8];
        v16bf A;
#pragma unroll
        for (int i = 0; i < 8; ++i) { A[i] = a0[i]; A[i + 8] = a1[i]; }

        // hoist all 12 B fragments (distinct regs) -> loads overlap, one wait
        v16bf Bt[12];
#pragma unroll
        for (int nt = 0; nt < 12; ++nt)
            Bt[nt] = *(const v16bf*)&Wbase[(size_t)nt * 16 * 768 + kb];

#pragma unroll
        for (int nt = 0; nt < 12; ++nt)
            acc[nt] = __builtin_amdgcn_wmma_f32_16x16x32_bf16(
                false, A, false, Bt[nt], (short)0, acc[nt], false, false);
    }

    // bias + per-row sum/sumsq over the full 1536-wide gate row
    float s1[8], s2[8];
#pragma unroll
    for (int r = 0; r < 8; ++r) { s1[r] = 0.f; s2[r] = 0.f; }
#pragma unroll
    for (int nt = 0; nt < 12; ++nt) {
        const int col = (wv * 12 + nt) * 16 + l15;
        const float bia = bih[col];
#pragma unroll
        for (int r = 0; r < 8; ++r) {
            float z = acc[nt][r] + bia;
            acc[nt][r] = z;
            s1[r] += z; s2[r] += z * z;
        }
    }
#pragma unroll
    for (int m = 1; m < 16; m <<= 1)
#pragma unroll
        for (int r = 0; r < 8; ++r) {
            s1[r] += __shfl_xor(s1[r], m, 32);
            s2[r] += __shfl_xor(s2[r], m, 32);
        }
    if (l15 == 0)
#pragma unroll
        for (int r = 0; r < 8; ++r) {
            rp[wv][hs * 8 + r]  = s1[r];
            rp2[wv][hs * 8 + r] = s2[r];
        }
    __syncthreads();

    float mean[8], rstd[8];
#pragma unroll
    for (int r = 0; r < 8; ++r) {
        const int row = hs * 8 + r;
        float su = 0.f, sq = 0.f;
#pragma unroll
        for (int w = 0; w < 8; ++w) { su += rp[w][row]; sq += rp2[w][row]; }
        const float mu = su * (1.0f / 1536.0f);
        const float va = sq * (1.0f / 1536.0f) - mu * mu;
        mean[r] = mu; rstd[r] = rsqrtf(va + EPSV);
    }

#pragma unroll
    for (int nt = 0; nt < 12; ++nt) {
        const int col = (wv * 12 + nt) * 16 + l15;
        const float gv = gg[col], bv = be[col];
#pragma unroll
        for (int r = 0; r < 8; ++r) {
            const int row = hs * 8 + r;
            float o = (acc[nt][r] - mean[r]) * rstd[r] * gv + bv;
            o_s[row * 1544 + col] = (bf16)o;
        }
    }
    __syncthreads();

    for (int idx = tid; idx < 16 * 192; idx += 256) {
        int r = idx / 192, c = idx % 192;
        *(uint4*)&Gx[(size_t)(m0 + r) * 1536 + c * 8] =
            *(const uint4*)&o_s[r * 1544 + c * 8];
    }
}

// ---------------------------------------------------------------------------
// Phase 2: persistent recurrent kernel.  grid = 8 (dir*4 + batch-tile).
// Per step: Gh = h@Whh^T (WMMA, A from LDS), LN over 1536, + Gx[t], gates,
// c = f*c + i*g, h = o*tanh(LN(c)), per-row logit partial (h . wo_half).
// ---------------------------------------------------------------------------
__global__ void __launch_bounds__(256)
k_recurrent(const bf16* __restrict__ GxF, const bf16* __restrict__ GxB,
            const bf16* __restrict__ WhhF, const bf16* __restrict__ WhhB,
            const float* __restrict__ bhhF, const float* __restrict__ bhhB,
            const float* __restrict__ ghhF, const float* __restrict__ behhF,
            const float* __restrict__ ghhB, const float* __restrict__ behhB,
            const float* __restrict__ ghoF, const float* __restrict__ behoF,
            const float* __restrict__ ghoB, const float* __restrict__ behoB,
            const float* __restrict__ wo_w,
            float* __restrict__ logitF, float* __restrict__ logitB)
{
    const int dir = blockIdx.x >> 2;
    const int b0  = (blockIdx.x & 3) * 16;
    const bf16*  Gx   = dir ? GxB  : GxF;
    const bf16*  Whh  = dir ? WhhB : WhhF;
    const float* bhh  = dir ? bhhB : bhhF;
    const float* gh   = dir ? ghhB : ghhF;
    const float* beh  = dir ? behhB : behhF;
    const float* gc   = dir ? ghoB : ghoF;
    const float* bec  = dir ? behoB : behoF;
    const float* whalf = wo_w + (dir ? 384 : 0);
    float* logit = dir ? logitB : logitF;

    const int tid = threadIdx.x;
    const int lane = tid & 31, wv = tid >> 5, hs = lane >> 4, l15 = lane & 15;
    const int gtype = wv >> 1;  // 0:i 1:f 2:o 3:g  (each type = 2 waves = 384 cols)

    __shared__ bf16  h_s[16 * 400];          // bf16 hidden state, padded stride
    __shared__ float c_s[16 * 384];          // fp32 cell state
    __shared__ bf16  act_s[4][16 * 384];     // activated gates i,f,o,g
    __shared__ float rp[8][16], rp2[8][16];  // gate-LN partials per wave
    __shared__ float cp[16][16], cp2[16][16];// cell-LN partials per thread-col
    __shared__ float lp_s[16][16];           // logit partials

    for (int i = tid; i < 16 * 400; i += 256) h_s[i] = (bf16)0.0f;
    for (int i = tid; i < 16 * 384; i += 256) c_s[i] = 0.0f;
    __syncthreads();

    // per-lane base pointer into Whh for this wave's 12 N-tiles
    const bf16* Wbase = Whh + (size_t)((wv * 12) * 16 + l15) * 384 + (hs ? 16 : 0);

    for (int step = 0; step < 512; ++step) {
        const int t = dir ? (511 - step) : step;

        // ---- Gh = h @ Whh^T  (16 x 1536, K = 384) ----
        v8f acc[12];
#pragma unroll
        for (int nt = 0; nt < 12; ++nt)
#pragma unroll
            for (int i = 0; i < 8; ++i) acc[nt][i] = 0.0f;

        for (int kt = 0; kt < 12; ++kt) {
            const int kb = kt * 32;
            v8bf a0 = *(const v8bf*)&h_s[l15 * 400 + kb + (hs ? 16 : 0)];
            v8bf a1 = *(const v8bf*)&h_s[l15 * 400 + kb + (hs ? 16 : 0) + 8];
            v16bf A;
#pragma unroll
            for (int i = 0; i < 8; ++i) { A[i] = a0[i]; A[i + 8] = a1[i]; }

            v16bf Bt[12];
#pragma unroll
            for (int nt = 0; nt < 12; ++nt)
                Bt[nt] = *(const v16bf*)&Wbase[(size_t)nt * 16 * 384 + kb];

#pragma unroll
            for (int nt = 0; nt < 12; ++nt)
                acc[nt] = __builtin_amdgcn_wmma_f32_16x16x32_bf16(
                    false, A, false, Bt[nt], (short)0, acc[nt], false, false);
        }

        // ---- bias + row reductions for LN over 1536 ----
        float s1[8], s2[8];
#pragma unroll
        for (int r = 0; r < 8; ++r) { s1[r] = 0.f; s2[r] = 0.f; }
#pragma unroll
        for (int nt = 0; nt < 12; ++nt) {
            const int col = (wv * 12 + nt) * 16 + l15;
            const float bia = bhh[col];
#pragma unroll
            for (int r = 0; r < 8; ++r) {
                float z = acc[nt][r] + bia;
                acc[nt][r] = z;
                s1[r] += z; s2[r] += z * z;
            }
        }
#pragma unroll
        for (int m = 1; m < 16; m <<= 1)
#pragma unroll
            for (int r = 0; r < 8; ++r) {
                s1[r] += __shfl_xor(s1[r], m, 32);
                s2[r] += __shfl_xor(s2[r], m, 32);
            }
        if (l15 == 0)
#pragma unroll
            for (int r = 0; r < 8; ++r) {
                rp[wv][hs * 8 + r]  = s1[r];
                rp2[wv][hs * 8 + r] = s2[r];
            }
        __syncthreads();   // B1

        float mean[8], rstd[8];
#pragma unroll
        for (int r = 0; r < 8; ++r) {
            const int row = hs * 8 + r;
            float su = 0.f, sq = 0.f;
#pragma unroll
            for (int w = 0; w < 8; ++w) { su += rp[w][row]; sq += rp2[w][row]; }
            const float mu = su * (1.0f / 1536.0f);
            const float va = sq * (1.0f / 1536.0f) - mu * mu;
            mean[r] = mu; rstd[r] = rsqrtf(va + EPSV);
        }

        // ---- normalize + add Gx[t] + nonlinearity, store to act_s ----
#pragma unroll
        for (int nt = 0; nt < 12; ++nt) {
            const int col = (wv * 12 + nt) * 16 + l15;
            const float gv = gh[col], bv = beh[col];
            const int jc = col - gtype * 384;
#pragma unroll
            for (int r = 0; r < 8; ++r) {
                const int row = hs * 8 + r;
                float z = (acc[nt][r] - mean[r]) * rstd[r] * gv + bv;
                float gx = (float)Gx[((size_t)(b0 + row) * 512 + t) * 1536 + col];
                float gate = z + gx;
                float av = (gtype == 3) ? tanhf(gate) : sigf(gate);
                act_s[gtype][row * 384 + jc] = (bf16)av;
            }
        }
        __syncthreads();   // B2

        // ---- cell update: c = f*c + i*g ----
        {
            const int row = tid >> 4, j0 = tid & 15;
            float sc = 0.f, sc2 = 0.f;
#pragma unroll
            for (int k = 0; k < 24; ++k) {
                const int j = j0 + k * 16;
                float iv = (float)act_s[0][row * 384 + j];
                float fv = (float)act_s[1][row * 384 + j];
                float gv2 = (float)act_s[3][row * 384 + j];
                float cn = fv * c_s[row * 384 + j] + iv * gv2;
                c_s[row * 384 + j] = cn;
                sc += cn; sc2 += cn * cn;
            }
            cp[row][j0] = sc; cp2[row][j0] = sc2;
        }
        __syncthreads();   // B3

        // ---- h = o * tanh(LN(c));  logit partial ----
        {
            const int row = tid >> 4, j0 = tid & 15;
            float su = 0.f, sq = 0.f;
#pragma unroll
            for (int q = 0; q < 16; ++q) { su += cp[row][q]; sq += cp2[row][q]; }
            const float mu = su * (1.0f / 384.0f);
            const float va = sq * (1.0f / 384.0f) - mu * mu;
            const float rs = rsqrtf(va + EPSV);
            float lp = 0.f;
#pragma unroll
            for (int k = 0; k < 24; ++k) {
                const int j = j0 + k * 16;
                float cn = c_s[row * 384 + j];
                float cl = (cn - mu) * rs * gc[j] + bec[j];
                float ov = (float)act_s[2][row * 384 + j];
                float hv = ov * tanhf(cl);
                h_s[row * 400 + j] = (bf16)hv;
                lp += hv * whalf[j];
            }
            lp_s[row][j0] = lp;
        }
        __syncthreads();   // B4

        if (tid < 16) {
            float su = 0.f;
#pragma unroll
            for (int q = 0; q < 16; ++q) su += lp_s[tid][q];
            logit[(size_t)(b0 + tid) * 512 + t] = su;
        }
        // next iteration's GEMM only reads h_s (stable since B4); lp_s is
        // rewritten only after B3 of next step, so no extra barrier needed.
    }
}

// ---------------------------------------------------------------------------
// Final: scores = sigmoid(logit_f + logit_b + x.wo_w + wo_b) * mask
// one wave per (b,t) row;  grid 4096 x 256
// ---------------------------------------------------------------------------
__global__ void __launch_bounds__(256)
k_final(const float* __restrict__ x, const float* __restrict__ mask,
        const float* __restrict__ wo_w, const float* __restrict__ wo_b,
        const float* __restrict__ logitF, const float* __restrict__ logitB,
        float* __restrict__ out)
{
    const int m = blockIdx.x * 8 + (threadIdx.x >> 5);
    const int lane = threadIdx.x & 31;
    const float* xr = x + (size_t)m * 768;
    float p = 0.f;
    for (int k = lane; k < 768; k += 32) p += xr[k] * wo_w[k];
#pragma unroll
    for (int s = 16; s > 0; s >>= 1) p += __shfl_xor(p, s, 32);
    if (lane == 0) {
        float l = p + logitF[m] + logitB[m] + wo_b[0];
        out[m] = sigf(l) * mask[m];
    }
}

// ---------------------------------------------------------------------------
extern "C" void kernel_launch(void* const* d_in, const int* in_sizes, int n_in,
                              void* d_out, int out_size, void* d_ws, size_t ws_size,
                              hipStream_t stream) {
    const float* x      = (const float*)d_in[0];
    const float* mask   = (const float*)d_in[1];
    const float* Wih_f  = (const float*)d_in[2];
    const float* bih_f  = (const float*)d_in[3];
    const float* Whh_f  = (const float*)d_in[4];
    const float* bhh_f  = (const float*)d_in[5];
    const float* gih_f  = (const float*)d_in[6];
    const float* beih_f = (const float*)d_in[7];
    const float* ghh_f  = (const float*)d_in[8];
    const float* behh_f = (const float*)d_in[9];
    const float* gho_f  = (const float*)d_in[10];
    const float* beho_f = (const float*)d_in[11];
    const float* Wih_b  = (const float*)d_in[12];
    const float* bih_b  = (const float*)d_in[13];
    const float* Whh_b  = (const float*)d_in[14];
    const float* bhh_b  = (const float*)d_in[15];
    const float* gih_b  = (const float*)d_in[16];
    const float* beih_b = (const float*)d_in[17];
    const float* ghh_b  = (const float*)d_in[18];
    const float* behh_b = (const float*)d_in[19];
    const float* gho_b  = (const float*)d_in[20];
    const float* beho_b = (const float*)d_in[21];
    const float* wo_w   = (const float*)d_in[22];
    const float* wo_b   = (const float*)d_in[23];

    const size_t NX   = (size_t)32768 * 768;   // x elements
    const size_t NWIH = (size_t)1536 * 768;
    const size_t NWHH = (size_t)1536 * 384;
    const size_t NGX  = (size_t)32768 * 1536;

    char* ws = (char*)d_ws;
    bf16* xb    = (bf16*)ws;  ws += NX * 2;
    bf16* WihFb = (bf16*)ws;  ws += NWIH * 2;
    bf16* WihBb = (bf16*)ws;  ws += NWIH * 2;
    bf16* WhhFb = (bf16*)ws;  ws += NWHH * 2;
    bf16* WhhBb = (bf16*)ws;  ws += NWHH * 2;
    bf16* GxF   = (bf16*)ws;  ws += NGX * 2;
    bf16* GxB   = (bf16*)ws;  ws += NGX * 2;
    float* lF   = (float*)ws; ws += (size_t)32768 * 4;
    float* lB   = (float*)ws; ws += (size_t)32768 * 4;

    k_cvt<<<2048, 256, 0, stream>>>(x,     xb,    (int)NX);
    k_cvt<<<512,  256, 0, stream>>>(Wih_f, WihFb, (int)NWIH);
    k_cvt<<<512,  256, 0, stream>>>(Wih_b, WihBb, (int)NWIH);
    k_cvt<<<256,  256, 0, stream>>>(Whh_f, WhhFb, (int)NWHH);
    k_cvt<<<256,  256, 0, stream>>>(Whh_b, WhhBb, (int)NWHH);

    dim3 g1(2048, 2, 1);
    k_inproj<<<g1, 256, 0, stream>>>(xb, WihFb, WihBb, bih_f, bih_b,
                                     gih_f, beih_f, gih_b, beih_b, GxF, GxB);

    k_recurrent<<<8, 256, 0, stream>>>(GxF, GxB, WhhFb, WhhBb,
                                       bhh_f, bhh_b, ghh_f, behh_f, ghh_b, behh_b,
                                       gho_f, beho_f, gho_b, beho_b,
                                       wo_w, lF, lB);

    k_final<<<4096, 256, 0, stream>>>(x, mask, wo_w, wo_b, lF, lB, (float*)d_out);
}
